// SelfAttention_47553877901789
// MI455X (gfx1250) — matrix operand order repaired and checked by reading, not compiled, not run on hardware
//
#include <hip/hip_runtime.h>
#include <hip/hip_bf16.h>

typedef _Float16 half_t;
typedef __attribute__((ext_vector_type(16))) _Float16 v16h;
typedef __attribute__((ext_vector_type(8)))  _Float16 v8h;
typedef __attribute__((ext_vector_type(8)))  float    v8f;
typedef __attribute__((ext_vector_type(4)))  int      v4i;

#define BATCH 2
#define SEQ   2048
#define DMODEL 1024
#define NHEAD 16
#define DHEAD 64
#define NTOK  (BATCH * SEQ)          // 4096
#define LOG2E 1.4426950408889634f

union V16H { v16h v; v8h h[2]; };

#ifndef __has_builtin
#define __has_builtin(x) 0
#endif
#if __has_builtin(__builtin_amdgcn_global_load_async_to_lds_b128) && \
    __has_builtin(__builtin_amdgcn_s_wait_asynccnt)
#define HAVE_ASYNC_LDS 1
#else
#define HAVE_ASYNC_LDS 0
#endif

// 16-byte global -> LDS copy; async (ASYNCcnt) path on gfx1250 toolchains
// that expose the builtin, synchronous fallback otherwise.
__device__ __forceinline__ void copy16_g2l(const half_t* g, half_t* l) {
#if HAVE_ASYNC_LDS
    __builtin_amdgcn_global_load_async_to_lds_b128(
        (__attribute__((address_space(1))) v4i*)g,
        (__attribute__((address_space(3))) v4i*)l, 0, 0);
#else
    *(v8h*)l = *(const v8h*)g;
#endif
}
__device__ __forceinline__ void wait_async_copies() {
#if HAVE_ASYNC_LDS
    __builtin_amdgcn_s_wait_asynccnt(0);
#endif
}

// ---------------------------------------------------------------------------
// f32 -> f16 weight conversion
// ---------------------------------------------------------------------------
__global__ void cvt_kernel(const float* __restrict__ w, half_t* __restrict__ o, int n) {
    int i = blockIdx.x * 256 + threadIdx.x;
    if (i < n) o[i] = (half_t)w[i];
}

// ---------------------------------------------------------------------------
// LayerNorm over D=1024, one block (256 threads) per token; f16 output
// ---------------------------------------------------------------------------
__global__ void __launch_bounds__(256) ln_kernel(const float* __restrict__ x,
                                                 const float* __restrict__ gamma,
                                                 const float* __restrict__ beta,
                                                 half_t* __restrict__ xn) {
    __shared__ float red[256];
    const int row = blockIdx.x;
    const int tid = threadIdx.x;
    const float* xr = x + (size_t)row * DMODEL;

    float s = 0.f, ss = 0.f;
    for (int i = tid; i < DMODEL; i += 256) { float v = xr[i]; s += v; ss += v * v; }

    red[tid] = s; __syncthreads();
    for (int off = 128; off > 0; off >>= 1) {
        if (tid < off) red[tid] += red[tid + off];
        __syncthreads();
    }
    const float mean = red[0] * (1.f / DMODEL);
    __syncthreads();
    red[tid] = ss; __syncthreads();
    for (int off = 128; off > 0; off >>= 1) {
        if (tid < off) red[tid] += red[tid + off];
        __syncthreads();
    }
    const float var  = red[0] * (1.f / DMODEL) - mean * mean;
    const float rstd = rsqrtf(var + 1e-5f);

    for (int i = tid; i < DMODEL; i += 256) {
        float v = (xr[i] - mean) * rstd * gamma[i] + beta[i];
        xn[(size_t)row * DMODEL + i] = (half_t)v;
    }
}

// ---------------------------------------------------------------------------
// GEMM: out[M x 1024] = A[M x 1024](f16) @ W[1024 x 1024](f16) + bias(f32)
// Block tile 128x128, 8 waves, each wave 32x64 (2x4 WMMA accumulators).
// ---------------------------------------------------------------------------
__global__ void __launch_bounds__(256)
gemm_kernel(const half_t* __restrict__ A, const half_t* __restrict__ W,
            const float* __restrict__ bias,
            half_t* __restrict__ outH, float* __restrict__ outF, int storeF32) {
    __shared__ __align__(16) half_t ldsB[128 * 64];      // [n][k], 16 KB

    const int lane = threadIdx.x & 31;
    const int wave = threadIdx.x >> 5;
    const int wm   = wave >> 1;        // 0..3 -> M offset
    const int wn   = wave & 1;         // 0..1 -> N offset
    const int g    = lane >> 4;        // half-wave group
    const int ln   = lane & 15;
    const int mBlock = blockIdx.y * 128;
    const int nBlock = blockIdx.x * 128;

    v8f acc[2][4];
#pragma unroll
    for (int ms = 0; ms < 2; ++ms)
#pragma unroll
        for (int ns = 0; ns < 4; ++ns) acc[ms][ns] = (v8f){};

    const int tn = threadIdx.x & 15;   // n-group of 8 cols
    const int tk = threadIdx.x >> 4;   // k row within pass

    for (int k0 = 0; k0 < DMODEL; k0 += 64) {
        __syncthreads();
        // stage W[k0:k0+64][nBlock:nBlock+128] transposed into ldsB[n][k]
#pragma unroll
        for (int pass = 0; pass < 4; ++pass) {
            const int kk = pass * 16 + tk;
            v8h wv = *(const v8h*)(W + (size_t)(k0 + kk) * DMODEL + nBlock + tn * 8);
#pragma unroll
            for (int j = 0; j < 8; ++j) ldsB[(tn * 8 + j) * 64 + kk] = wv[j];
        }
        // prefetch next W tile into cache while we compute on this one
        if (k0 + 64 < DMODEL)
            __builtin_prefetch(W + (size_t)(k0 + 64 + tk) * DMODEL + nBlock + tn * 8, 0, 0);
        __syncthreads();

#pragma unroll
        for (int kc = 0; kc < 2; ++kc) {
            v16h a[2], b[4];
#pragma unroll
            for (int ms = 0; ms < 2; ++ms) {
                const half_t* p = A + (size_t)(mBlock + wm * 32 + ms * 16 + ln) * DMODEL
                                    + k0 + kc * 32 + g * 8;
                V16H u; u.h[0] = *(const v8h*)p; u.h[1] = *(const v8h*)(p + 16);
                a[ms] = u.v;
            }
#pragma unroll
            for (int ns = 0; ns < 4; ++ns) {
                const half_t* p = &ldsB[(wn * 64 + ns * 16 + ln) * 64 + kc * 32 + g * 8];
                V16H u; u.h[0] = *(const v8h*)p; u.h[1] = *(const v8h*)(p + 16);
                b[ns] = u.v;
            }
#pragma unroll
            for (int ms = 0; ms < 2; ++ms)
#pragma unroll
                for (int ns = 0; ns < 4; ++ns)
                    acc[ms][ns] = __builtin_amdgcn_wmma_f32_16x16x32_f16(
                        false, a[ms], false, b[ns], (short)0, acc[ms][ns], false, false);
        }
    }

    // epilogue: bias + store (C layout: vgpr i -> row i + g*8, col = ln)
#pragma unroll
    for (int ms = 0; ms < 2; ++ms)
#pragma unroll
        for (int ns = 0; ns < 4; ++ns) {
            const int col = nBlock + wn * 64 + ns * 16 + ln;
            const float bv = bias[col];
#pragma unroll
            for (int i = 0; i < 8; ++i) {
                const int row = mBlock + wm * 32 + ms * 16 + g * 8 + i;
                const float v = acc[ms][ns][i] + bv;
                if (storeF32) outF[(size_t)row * DMODEL + col] = v;
                else          outH[(size_t)row * DMODEL + col] = (half_t)v;
            }
        }
}

// ---------------------------------------------------------------------------
// Flash attention: one block per (b, h, 128-query tile); 8 waves x 16 rows.
// K tile staged to LDS via async global->LDS b128 copies (ASYNCcnt path);
// V tile staged transposed so P*V B-frags are contiguous ds_load_b128s.
// Online softmax in WMMA C-layout; P reshaped through per-wave LDS tile.
// ---------------------------------------------------------------------------
__global__ void __launch_bounds__(256)
attn_kernel(const half_t* __restrict__ Q, const half_t* __restrict__ K,
            const half_t* __restrict__ V, half_t* __restrict__ O) {
    __shared__ __align__(16) half_t ldsK [64 * 64];      // [key][dh]   8 KB
    __shared__ __align__(16) half_t ldsVt[64 * 64];      // [dh][key]   8 KB
    __shared__ __align__(16) half_t ldsP [8][16 * 64];   // per-wave P 16 KB

    const int tid  = threadIdx.x;
    const int lane = tid & 31;
    const int wave = tid >> 5;
    const int g    = lane >> 4;
    const int ln   = lane & 15;

    const int qt  = blockIdx.x % (SEQ / 128);
    const int bh  = blockIdx.x / (SEQ / 128);
    const int h   = bh % NHEAD;
    const int b   = bh / NHEAD;
    const int qrow0 = qt * 128 + wave * 16;

    const half_t* qh = Q + (size_t)b * SEQ * DMODEL + h * DHEAD;
    const half_t* kh = K + (size_t)b * SEQ * DMODEL + h * DHEAD;
    const half_t* vh = V + (size_t)b * SEQ * DMODEL + h * DHEAD;

    // Q A-fragments (fixed across key blocks): dh chunks of 32
    v16h qf[2];
#pragma unroll
    for (int c = 0; c < 2; ++c) {
        const half_t* p = qh + (size_t)(qrow0 + ln) * DMODEL + c * 32 + g * 8;
        V16H u; u.h[0] = *(const v8h*)p; u.h[1] = *(const v8h*)(p + 16);
        qf[c] = u.v;
    }

    float m[8], l[8];
    v8f o[4];
#pragma unroll
    for (int i = 0; i < 8; ++i) { m[i] = -1e30f; l[i] = 0.f; }
#pragma unroll
    for (int t = 0; t < 4; ++t) o[t] = (v8f){};

    for (int kb = 0; kb < SEQ; kb += 64) {
        __syncthreads();   // all waves done reading previous K/V tiles

        // ---- stage K tile [64 x 64] row-major via async b128 copies ----
#pragma unroll
        for (int pass = 0; pass < 2; ++pass) {
            const int c   = pass * 256 + tid;       // 512 x 16B chunks
            const int row = c >> 3;
            const int col = (c & 7) * 8;
            copy16_g2l(kh + (size_t)(kb + row) * DMODEL + col, &ldsK[row * 64 + col]);
        }
        // ---- stage V tile transposed [dh][key] (register transpose) ----
#pragma unroll
        for (int pass = 0; pass < 2; ++pass) {
            const int c   = pass * 256 + tid;
            const int key = c >> 3;
            const int dg  = (c & 7) * 8;
            v8h vv = *(const v8h*)(vh + (size_t)(kb + key) * DMODEL + dg);
#pragma unroll
            for (int j = 0; j < 8; ++j) ldsVt[(dg + j) * 64 + key] = vv[j];
        }
        wait_async_copies();
        __syncthreads();

        // ---- scores: S = Q K^T (K^T B-frags contiguous in ldsK) ----
        v8f s[4];
#pragma unroll
        for (int t = 0; t < 4; ++t) {
            v8f c = (v8f){};
#pragma unroll
            for (int ch = 0; ch < 2; ++ch) {
                const half_t* p = &ldsK[(t * 16 + ln) * 64 + ch * 32 + g * 8];
                V16H u; u.h[0] = *(const v8h*)p; u.h[1] = *(const v8h*)(p + 16);
                c = __builtin_amdgcn_wmma_f32_16x16x32_f16(
                        false, qf[ch], false, u.v, (short)0, c, false, false);
            }
            s[t] = c;
        }

        // ---- online softmax (scale = 1/sqrt(64) = 0.125) ----
#pragma unroll
        for (int i = 0; i < 8; ++i) {
            float v = fmaxf(fmaxf(s[0][i], s[1][i]), fmaxf(s[2][i], s[3][i]));
            v = fmaxf(v, __shfl_xor(v, 1, 32));
            v = fmaxf(v, __shfl_xor(v, 2, 32));
            v = fmaxf(v, __shfl_xor(v, 4, 32));
            v = fmaxf(v, __shfl_xor(v, 8, 32));
            const float mn  = fmaxf(m[i], v * 0.125f);
            const float fac = exp2f((m[i] - mn) * LOG2E);
            float rs = 0.f;
#pragma unroll
            for (int t = 0; t < 4; ++t) {
                const float pp = exp2f((s[t][i] * 0.125f - mn) * LOG2E);
                s[t][i] = pp;
                rs += pp;
            }
            rs += __shfl_xor(rs, 1, 32);
            rs += __shfl_xor(rs, 2, 32);
            rs += __shfl_xor(rs, 4, 32);
            rs += __shfl_xor(rs, 8, 32);
            l[i] = l[i] * fac + rs;
            m[i] = mn;
#pragma unroll
            for (int t = 0; t < 4; ++t) o[t][i] *= fac;
        }

        // ---- P (C-layout) -> per-wave LDS (row-major 16x64 f16) ----
#pragma unroll
        for (int t = 0; t < 4; ++t)
#pragma unroll
            for (int i = 0; i < 8; ++i)
                ldsP[wave][(g * 8 + i) * 64 + t * 16 + ln] = (half_t)s[t][i];
        // per-wave LDS region: DS ops are in-order within a wave, no barrier

        // ---- O += P V (V B-frags contiguous in ldsVt) ----
#pragma unroll
        for (int ch = 0; ch < 2; ++ch) {
            v16h pf;
            {
                const half_t* p = &ldsP[wave][ln * 64 + ch * 32 + g * 8];
                V16H u; u.h[0] = *(const v8h*)p; u.h[1] = *(const v8h*)(p + 16);
                pf = u.v;
            }
#pragma unroll
            for (int t = 0; t < 4; ++t) {
                const half_t* p = &ldsVt[(t * 16 + ln) * 64 + ch * 32 + g * 8];
                V16H u; u.h[0] = *(const v8h*)p; u.h[1] = *(const v8h*)(p + 16);
                o[t] = __builtin_amdgcn_wmma_f32_16x16x32_f16(
                            false, pf, false, u.v, (short)0, o[t], false, false);
            }
        }
    }

    // ---- normalize and store attention output (f16, [b, s, h, dh]) ----
    float inv[8];
#pragma unroll
    for (int i = 0; i < 8; ++i) inv[i] = 1.f / l[i];
#pragma unroll
    for (int t = 0; t < 4; ++t)
#pragma unroll
        for (int i = 0; i < 8; ++i) {
            const int row = qrow0 + g * 8 + i;
            O[((size_t)b * SEQ + row) * DMODEL + h * DHEAD + t * 16 + ln] =
                (half_t)(o[t][i] * inv[i]);
        }
}

// ---------------------------------------------------------------------------
// Host launcher
// ---------------------------------------------------------------------------
extern "C" void kernel_launch(void* const* d_in, const int* in_sizes, int n_in,
                              void* d_out, int out_size, void* d_ws, size_t ws_size,
                              hipStream_t stream) {
    (void)in_sizes; (void)n_in; (void)out_size; (void)ws_size;

    const float* x    = (const float*)d_in[0];
    const float* ln_g = (const float*)d_in[1];
    const float* ln_b = (const float*)d_in[2];
    const float* wq   = (const float*)d_in[3];
    const float* bq   = (const float*)d_in[4];
    const float* wk   = (const float*)d_in[5];
    const float* bk   = (const float*)d_in[6];
    const float* wv   = (const float*)d_in[7];
    const float* bv   = (const float*)d_in[8];
    const float* wo   = (const float*)d_in[9];
    const float* bo   = (const float*)d_in[10];
    float* out = (float*)d_out;

    char* ws = (char*)d_ws;
    size_t off = 0;
    const size_t actBytes = (size_t)NTOK * DMODEL * sizeof(half_t);   // 8 MiB
    const size_t wBytes   = (size_t)DMODEL * DMODEL * sizeof(half_t); // 2 MiB

    half_t* xn  = (half_t*)(ws + off); off += actBytes;
    half_t* wqh = (half_t*)(ws + off); off += wBytes;
    half_t* wkh = (half_t*)(ws + off); off += wBytes;
    half_t* wvh = (half_t*)(ws + off); off += wBytes;
    half_t* woh = (half_t*)(ws + off); off += wBytes;
    half_t* Qb  = (half_t*)(ws + off); off += actBytes;
    half_t* Kb  = (half_t*)(ws + off); off += actBytes;
    half_t* Vb  = (half_t*)(ws + off); off += actBytes;
    half_t* Ah  = (half_t*)(ws + off); off += actBytes;

    const int wN = DMODEL * DMODEL;
    cvt_kernel<<<(wN + 255) / 256, 256, 0, stream>>>(wq, wqh, wN);
    cvt_kernel<<<(wN + 255) / 256, 256, 0, stream>>>(wk, wkh, wN);
    cvt_kernel<<<(wN + 255) / 256, 256, 0, stream>>>(wv, wvh, wN);
    cvt_kernel<<<(wN + 255) / 256, 256, 0, stream>>>(wo, woh, wN);

    ln_kernel<<<NTOK, 256, 0, stream>>>(x, ln_g, ln_b, xn);

    dim3 gg(DMODEL / 128, NTOK / 128);   // (8, 32)
    gemm_kernel<<<gg, 256, 0, stream>>>(xn, wqh, bq, Qb, nullptr, 0);
    gemm_kernel<<<gg, 256, 0, stream>>>(xn, wkh, bk, Kb, nullptr, 0);
    gemm_kernel<<<gg, 256, 0, stream>>>(xn, wvh, bv, Vb, nullptr, 0);

    attn_kernel<<<BATCH * NHEAD * (SEQ / 128), 256, 0, stream>>>(Qb, Kb, Vb, Ah);

    gemm_kernel<<<gg, 256, 0, stream>>>(Ah, woh, bo, nullptr, out, 1);
}
